// GPTQBigCodeModel_68195490726270
// MI455X (gfx1250) — matrix operand order repaired and checked by reading, not compiled, not run on hardware
//
#include <hip/hip_runtime.h>
#include <hip/hip_bf16.h>
#include <math.h>

// ---------------- model constants ----------------
#define NLAYER 8
#define BB     4
#define SS     128
#define DD     2048
#define HH     16
#define HDIM   128
#define FF     8192
#define VV     49152
#define GRPQ   128
#define LNEPS  1e-5f
#define NQKV   (DD + 2 * HDIM)   // 2304
#define MROWS  (BB * SS)         // 512

// ---------------- GEMM tiling ----------------
#define BM 128
#define BN 128
#define BK 32
#define LDSTR 40                 // halves per LDS tile row (pad keeps 16B align)

typedef _Float16 v16h __attribute__((ext_vector_type(16)));
typedef _Float16 h8   __attribute__((ext_vector_type(8)));
typedef float    v8f  __attribute__((ext_vector_type(8)));
typedef float    f4   __attribute__((ext_vector_type(4)));
typedef int      i4v  __attribute__((vector_size(16)));   // matches async-LDS builtin pointee

// ---- CDNA5 async global->LDS staging (ASYNCcnt), guarded for toolchain ----
#if __has_builtin(__builtin_amdgcn_global_load_async_to_lds_b128) && \
    __has_builtin(__builtin_amdgcn_s_wait_asynccnt)
#define USE_ASYNC_LDS 1
#else
#define USE_ASYNC_LDS 0
#endif

// Stage 32 bytes (16 halves) of the A tile from global to LDS.
__device__ __forceinline__ void stage_a16(const _Float16* ga, _Float16* ldst) {
#if USE_ASYNC_LDS
    __builtin_amdgcn_global_load_async_to_lds_b128(
        (__attribute__((address_space(1))) i4v*)ga,
        (__attribute__((address_space(3))) i4v*)ldst, 0, 0);
    __builtin_amdgcn_global_load_async_to_lds_b128(
        (__attribute__((address_space(1))) i4v*)ga,
        (__attribute__((address_space(3))) i4v*)ldst, 16, 0);
#else
    h8 a0 = *(const h8*)ga;
    h8 a1 = *(const h8*)(ga + 8);
    *(h8*)ldst       = a0;
    *(h8*)(ldst + 8) = a1;
#endif
}

__device__ __forceinline__ void wait_async_stage() {
#if USE_ASYNC_LDS
    __builtin_amdgcn_s_wait_asynccnt(0);
#endif
}

// ---- fragment loaders (CDNA5 wave32 WMMA 16x16x32 f16 layouts) ----
// A (16x32, 16-bit): lanes 0-15 -> M=lane, halves [0..7]=K0..7, [8..15]=K16..23
//                    lanes 16-31 -> M=lane-16, halves [0..7]=K8..15, [8..15]=K24..31
__device__ __forceinline__ v16h frag_a(const _Float16* base) {
    int lane = threadIdx.x & 31;
    int r    = lane & 15;
    int kb   = (lane >> 4) << 3;          // 0 or 8
    const _Float16* p = base + (size_t)r * LDSTR + kb;
    h8 lo = *(const h8*)p;
    h8 hi = *(const h8*)(p + 16);
    v16h f;
#pragma unroll
    for (int i = 0; i < 8; ++i) { f[i] = lo[i]; f[i + 8] = hi[i]; }
    return f;
}
// B (32x16, 16-bit, stored N-major in LDS): lanes 0-15 -> N=lane, halves [0..7]=K0..7, [8..15]=K16..23
//                                           lanes 16-31 -> N=lane-16, halves [0..7]=K8..15, [8..15]=K24..31
__device__ __forceinline__ v16h frag_b(const _Float16* base) {
    int lane = threadIdx.x & 31;
    int c    = lane & 15;
    int kb   = (lane >> 4) << 3;          // 0 or 8
    const _Float16* p = base + (size_t)c * LDSTR + kb;
    h8 lo = *(const h8*)p;
    h8 hi = *(const h8*)(p + 16);
    v16h f;
#pragma unroll
    for (int i = 0; i < 8; ++i) { f[i] = lo[i]; f[i + 8] = hi[i]; }
    return f;
}

template <bool HAS_BIAS, bool HAS_RESID>
__device__ __forceinline__ void gemm_epilogue(v8f acc[2][4], const float* bias,
                                              const float* resid, float* C,
                                              int m0, int n0, int wm, int wn, int N) {
    int lane = threadIdx.x & 31;
    int cn   = lane & 15;
    int mh   = (lane >> 4) << 3;          // 0 or 8
#pragma unroll
    for (int i = 0; i < 2; ++i) {
#pragma unroll
        for (int j = 0; j < 4; ++j) {
            int mbase = m0 + wm * 32 + i * 16 + mh;
            int col   = n0 + wn * 64 + j * 16 + cn;
            float bv  = HAS_BIAS ? bias[col] : 0.0f;
#pragma unroll
            for (int r = 0; r < 8; ++r) {
                size_t idx = (size_t)(mbase + r) * N + col;
                float v = acc[i][j][r] + bv;
                if (HAS_RESID) v += resid[idx];
                C[idx] = v;
            }
        }
    }
}

// ---------------- fused GPTQ-dequant + WMMA GEMM ----------------
// C[M,N] = A_f16[M,K] @ dequant(qw,qz,sc)[K,N] (+ bias) (+ resid)
template <bool HAS_BIAS, bool HAS_RESID>
__global__ __launch_bounds__(256)
void gemm_deq_kernel(const _Float16* __restrict__ A,
                     const int* __restrict__ qw,   // [K/8, N] nibbles along K
                     const int* __restrict__ qz,   // [K/GRP, N/8] nibbles along N
                     const float* __restrict__ sc, // [K/GRP, N]
                     const float* __restrict__ bias,
                     const float* __restrict__ resid,
                     float* __restrict__ C,
                     int M, int N, int K) {
    __shared__ alignas(16) _Float16 As[BM * LDSTR];
    __shared__ alignas(16) _Float16 Bs[BN * LDSTR];
    int tid = threadIdx.x;
    int wid = tid >> 5;
    int wm  = wid & 3;                    // 4 wave-rows of 32
    int wn  = wid >> 2;                   // 2 wave-cols of 64
    int m0  = blockIdx.y * BM;
    int n0  = blockIdx.x * BN;

    v8f acc[2][4];
    v8f zer = {0.f,0.f,0.f,0.f,0.f,0.f,0.f,0.f};
#pragma unroll
    for (int i = 0; i < 2; ++i)
#pragma unroll
        for (int j = 0; j < 4; ++j) acc[i][j] = zer;

    int arow = tid >> 1;                  // 0..127
    int aseg = (tid & 1) << 4;            // 0 or 16 halves

    for (int k0 = 0; k0 < K; k0 += BK) {
        const _Float16* ga = A + (size_t)(m0 + arow) * K + k0 + aseg;
        int g = k0 / GRPQ;                // BK=32 divides GRP=128: one group per tile
        __syncthreads();
        // Kick the A-tile DMA (ASYNCcnt) so it overlaps the dequant VALU work.
        stage_a16(ga, &As[arow * LDSTR + aseg]);
        if (k0 + BK < K) __builtin_prefetch(ga + BK, 0, 0);   // global_prefetch_b8
#pragma unroll
        for (int t = 0; t < 2; ++t) {
            int task = tid + t * 256;     // 512 tasks: 128 cols x 4 k-octets
            int c    = task >> 2;
            int r    = task & 3;
            int n    = n0 + c;
            unsigned int q = (unsigned int)qw[(size_t)(k0 / 8 + r) * N + n];
            float s  = sc[(size_t)g * N + n];
            float zf = (float)(((((unsigned int)qz[(size_t)g * (N >> 3) + (n >> 3)])
                                 >> ((n & 7) * 4)) & 15u) + 1u);
            h8 wv;
#pragma unroll
            for (int j = 0; j < 8; ++j) {
                float w = (float)((q >> (4 * j)) & 15u) - zf;
                wv[j] = (_Float16)(w * s);
            }
            *(h8*)(&Bs[c * LDSTR + (r << 3)]) = wv;
        }
        wait_async_stage();
        __syncthreads();

        v16h af[2], bf[4];
#pragma unroll
        for (int i = 0; i < 2; ++i) af[i] = frag_a(&As[(wm * 32 + i * 16) * LDSTR]);
#pragma unroll
        for (int j = 0; j < 4; ++j) bf[j] = frag_b(&Bs[(wn * 64 + j * 16) * LDSTR]);
#pragma unroll
        for (int i = 0; i < 2; ++i)
#pragma unroll
            for (int j = 0; j < 4; ++j)
                acc[i][j] = __builtin_amdgcn_wmma_f32_16x16x32_f16(
                    false, af[i], false, bf[j], (short)0, acc[i][j], false, false);
    }
    gemm_epilogue<HAS_BIAS, HAS_RESID>(acc, bias, resid, C, m0, n0, wm, wn, N);
}

// ---------------- logits GEMM: C = A_f16 @ wte^T (wte fp32 [N,K]) ----------------
__global__ __launch_bounds__(256)
void gemm_wteT_kernel(const _Float16* __restrict__ A,
                      const float* __restrict__ Wt,  // [N, K] fp32, used transposed
                      float* __restrict__ C,
                      int M, int N, int K) {
    __shared__ alignas(16) _Float16 As[BM * LDSTR];
    __shared__ alignas(16) _Float16 Bs[BN * LDSTR];
    int tid = threadIdx.x;
    int wid = tid >> 5;
    int wm  = wid & 3;
    int wn  = wid >> 2;
    int m0  = blockIdx.y * BM;
    int n0  = blockIdx.x * BN;

    v8f acc[2][4];
    v8f zer = {0.f,0.f,0.f,0.f,0.f,0.f,0.f,0.f};
#pragma unroll
    for (int i = 0; i < 2; ++i)
#pragma unroll
        for (int j = 0; j < 4; ++j) acc[i][j] = zer;

    int arow = tid >> 1;
    int aseg = (tid & 1) << 4;

    for (int k0 = 0; k0 < K; k0 += BK) {
        const _Float16* ga = A + (size_t)(m0 + arow) * K + k0 + aseg;
        __syncthreads();
        stage_a16(ga, &As[arow * LDSTR + aseg]);
#pragma unroll
        for (int t = 0; t < 2; ++t) {
            int task = tid + t * 256;
            int c    = task >> 2;
            int r    = task & 3;
            const float* src = Wt + (size_t)(n0 + c) * K + k0 + r * 8;
            f4 x0 = *(const f4*)src;
            f4 x1 = *(const f4*)(src + 4);
            h8 wv;
#pragma unroll
            for (int j = 0; j < 4; ++j) { wv[j] = (_Float16)x0[j]; wv[4 + j] = (_Float16)x1[j]; }
            *(h8*)(&Bs[c * LDSTR + (r << 3)]) = wv;
        }
        wait_async_stage();
        __syncthreads();

        v16h af[2], bf[4];
#pragma unroll
        for (int i = 0; i < 2; ++i) af[i] = frag_a(&As[(wm * 32 + i * 16) * LDSTR]);
#pragma unroll
        for (int j = 0; j < 4; ++j) bf[j] = frag_b(&Bs[(wn * 64 + j * 16) * LDSTR]);
#pragma unroll
        for (int i = 0; i < 2; ++i)
#pragma unroll
            for (int j = 0; j < 4; ++j)
                acc[i][j] = __builtin_amdgcn_wmma_f32_16x16x32_f16(
                    false, af[i], false, bf[j], (short)0, acc[i][j], false, false);
    }
    gemm_epilogue<false, false>(acc, nullptr, nullptr, C, m0, n0, wm, wn, N);
}

// ---------------- embedding: h = wte[x] + wpe[pos] ----------------
__global__ __launch_bounds__(256)
void embed_kernel(const int* __restrict__ x, const float* __restrict__ wte,
                  const float* __restrict__ wpe, float* __restrict__ h) {
    int row = blockIdx.x;                 // b*SS + s
    int sp  = row & (SS - 1);
    int tok = x[row];
    const float* tw = wte + (size_t)tok * DD;
    const float* pw = wpe + (size_t)sp * DD;
    float* hr = h + (size_t)row * DD;
    for (int d = threadIdx.x; d < DD; d += 256) hr[d] = tw[d] + pw[d];
}

// ---------------- LayerNorm fp32 -> f16 activations ----------------
__global__ __launch_bounds__(256)
void ln_kernel(const float* __restrict__ xin, const float* __restrict__ w,
               const float* __restrict__ b, _Float16* __restrict__ out, int Dn) {
    int row = blockIdx.x;
    const float* xr = xin + (size_t)row * Dn;
    int tid = threadIdx.x;
    float s = 0.f, s2 = 0.f;
    for (int d = tid; d < Dn; d += 256) { float v = xr[d]; s += v; s2 += v * v; }
    __shared__ float r1[256];
    __shared__ float r2[256];
    r1[tid] = s; r2[tid] = s2;
    __syncthreads();
    for (int o = 128; o > 0; o >>= 1) {
        if (tid < o) { r1[tid] += r1[tid + o]; r2[tid] += r2[tid + o]; }
        __syncthreads();
    }
    float mean = r1[0] / (float)Dn;
    float var  = r2[0] / (float)Dn - mean * mean;
    float rstd = rsqrtf(var + LNEPS);
    _Float16* orow = out + (size_t)row * Dn;
    for (int d = tid; d < Dn; d += 256)
        orow[d] = (_Float16)((xr[d] - mean) * rstd * w[d] + b[d]);
}

// ---------------- tanh-GELU fp32 -> f16 ----------------
__global__ __launch_bounds__(256)
void gelu_kernel(const float* __restrict__ in, _Float16* __restrict__ out, int n) {
    int i = blockIdx.x * 256 + threadIdx.x;
    if (i < n) {
        float x = in[i];
        float t = tanhf(0.7978845608028654f * (x + 0.044715f * x * x * x));
        out[i] = (_Float16)(0.5f * x * (1.0f + t));
    }
}

// ---------------- MQA attention, causal, S=128, HD=128 ----------------
// block = (b,h), 128 threads = one query row each; writes f16 [512, D]
__global__ __launch_bounds__(128)
void attn_kernel(const float* __restrict__ qkv, _Float16* __restrict__ out) {
    __shared__ float scs[SS * SS];        // 64 KB score matrix
    int b    = blockIdx.x >> 4;
    int hh   = blockIdx.x & 15;
    int qpos = threadIdx.x;
    const float* qrow = qkv + (size_t)(b * SS + qpos) * NQKV + hh * HDIM;
    float qv[HDIM];
#pragma unroll
    for (int d = 0; d < HDIM; ++d) qv[d] = qrow[d];
    const float inv = 0.08838834764831845f;  // 1/sqrt(128)
    for (int kp = 0; kp < SS; ++kp) {
        const float* kr = qkv + (size_t)(b * SS + kp) * NQKV + DD;   // wave-uniform
        float dot = 0.f;
#pragma unroll 16
        for (int d = 0; d < HDIM; ++d) dot += qv[d] * kr[d];
        scs[qpos * SS + kp] = (kp <= qpos) ? dot * inv : -1e9f;
    }
    float mx = -3.4e38f;
    for (int kp = 0; kp < SS; ++kp) mx = fmaxf(mx, scs[qpos * SS + kp]);
    float sum = 0.f;
    for (int kp = 0; kp < SS; ++kp) {
        float e = __expf(scs[qpos * SS + kp] - mx);
        scs[qpos * SS + kp] = e;
        sum += e;
    }
    float rs = 1.f / sum;
#pragma unroll
    for (int d = 0; d < HDIM; ++d) qv[d] = 0.f;   // reuse as output accumulator
    for (int kp = 0; kp <= qpos; ++kp) {
        const float* vr = qkv + (size_t)(b * SS + kp) * NQKV + DD + HDIM;
        float p = scs[qpos * SS + kp] * rs;
#pragma unroll 16
        for (int d = 0; d < HDIM; ++d) qv[d] += p * vr[d];
    }
    _Float16* orow = out + (size_t)(b * SS + qpos) * DD + hh * HDIM;
#pragma unroll
    for (int d = 0; d < HDIM; ++d) orow[d] = (_Float16)qv[d];
}

// ---------------- host orchestration ----------------
extern "C" void kernel_launch(void* const* d_in, const int* in_sizes, int n_in,
                              void* d_out, int out_size, void* d_ws, size_t ws_size,
                              hipStream_t stream) {
    (void)in_sizes; (void)n_in; (void)out_size; (void)ws_size;
    const int*   x    = (const int*)d_in[0];
    const float* wte  = (const float*)d_in[1];
    const float* wpe  = (const float*)d_in[2];
    const float* ln1w = (const float*)d_in[3];
    const float* ln1b = (const float*)d_in[4];
    const float* ln2w = (const float*)d_in[5];
    const float* ln2b = (const float*)d_in[6];
    const float* lnfw = (const float*)d_in[7];
    const float* lnfb = (const float*)d_in[8];
    const int*   aqw  = (const int*)d_in[9];
    const int*   aqz  = (const int*)d_in[10];
    const float* asc  = (const float*)d_in[11];
    const float* abia = (const float*)d_in[12];
    const int*   pqw  = (const int*)d_in[13];
    const int*   pqz  = (const int*)d_in[14];
    const float* psc  = (const float*)d_in[15];
    const float* pbia = (const float*)d_in[16];
    const int*   fqw  = (const int*)d_in[17];
    const int*   fqz  = (const int*)d_in[18];
    const float* fsc  = (const float*)d_in[19];
    const float* fbia = (const float*)d_in[20];
    const int*   oqw  = (const int*)d_in[21];
    const int*   oqz  = (const int*)d_in[22];
    const float* osc  = (const float*)d_in[23];
    const float* obia = (const float*)d_in[24];

    char* ws = (char*)d_ws;
    float*    h   = (float*)(ws);                                      //  4 MB
    _Float16* a16 = (_Float16*)(ws + (size_t)4 * 1024 * 1024);         //  8 MB
    float*    qkv = (float*)(ws + (size_t)12 * 1024 * 1024);           //  4.5 MB
    float*    mfc = (float*)(ws + (size_t)17 * 1024 * 1024);           // 16 MB

    dim3 blk256(256), blk128(128);

    embed_kernel<<<MROWS, blk256, 0, stream>>>(x, wte, wpe, h);

    for (int l = 0; l < NLAYER; ++l) {
        // ---- attention ----
        ln_kernel<<<MROWS, blk256, 0, stream>>>(h, ln1w + (size_t)l * DD,
                                                ln1b + (size_t)l * DD, a16, DD);
        gemm_deq_kernel<true, false><<<dim3(NQKV / BN, MROWS / BM), blk256, 0, stream>>>(
            a16,
            aqw + (size_t)l * (DD / 8) * NQKV,
            aqz + (size_t)l * (DD / GRPQ) * (NQKV / 8),
            asc + (size_t)l * (DD / GRPQ) * NQKV,
            abia + (size_t)l * NQKV, nullptr, qkv, MROWS, NQKV, DD);
        attn_kernel<<<BB * HH, blk128, 0, stream>>>(qkv, a16);
        gemm_deq_kernel<true, true><<<dim3(DD / BN, MROWS / BM), blk256, 0, stream>>>(
            a16,
            pqw + (size_t)l * (DD / 8) * DD,
            pqz + (size_t)l * (DD / GRPQ) * (DD / 8),
            psc + (size_t)l * (DD / GRPQ) * DD,
            pbia + (size_t)l * DD, h, h, MROWS, DD, DD);
        // ---- MLP ----
        ln_kernel<<<MROWS, blk256, 0, stream>>>(h, ln2w + (size_t)l * DD,
                                                ln2b + (size_t)l * DD, a16, DD);
        gemm_deq_kernel<true, false><<<dim3(FF / BN, MROWS / BM), blk256, 0, stream>>>(
            a16,
            fqw + (size_t)l * (DD / 8) * FF,
            fqz + (size_t)l * (DD / GRPQ) * (FF / 8),
            fsc + (size_t)l * (DD / GRPQ) * FF,
            fbia + (size_t)l * FF, nullptr, mfc, MROWS, FF, DD);
        gelu_kernel<<<(MROWS * FF + 255) / 256, blk256, 0, stream>>>(mfc, a16, MROWS * FF);
        gemm_deq_kernel<true, true><<<dim3(DD / BN, MROWS / BM), blk256, 0, stream>>>(
            a16,
            oqw + (size_t)l * (FF / 8) * DD,
            oqz + (size_t)l * (FF / GRPQ) * (DD / 8),
            osc + (size_t)l * (FF / GRPQ) * DD,
            obia + (size_t)l * DD, h, h, MROWS, DD, FF);
    }

    ln_kernel<<<MROWS, blk256, 0, stream>>>(h, lnfw, lnfb, a16, DD);
    gemm_wteT_kernel<<<dim3(VV / BN, MROWS / BM), blk256, 0, stream>>>(
        a16, wte, (float*)d_out, MROWS, VV, DD);
}
